// HGATLayer_49246095016355
// MI455X (gfx1250) — compile-verified, matrix operand order checked by reading, not run on hardware
//
#include <hip/hip_runtime.h>
#include <hip/hip_bf16.h>
#include <math.h>

typedef __attribute__((ext_vector_type(16))) _Float16 v16h;
typedef __attribute__((ext_vector_type(8)))  float    v8f;

#define MAXN_PROJ (1.0f - 4e-3f)

// ---------------- helpers ----------------

__device__ __forceinline__ float block_sum_256(float v) {
  __shared__ float red[256];
  int t = threadIdx.x;
  red[t] = v; __syncthreads();
  #pragma unroll
  for (int s = 128; s > 0; s >>= 1) {
    if (t < s) red[t] += red[t + s];
    __syncthreads();
  }
  float r = red[0];
  __syncthreads();
  return r;
}

// order-preserving float -> uint encoding (monotone for all finite floats)
__device__ __forceinline__ unsigned enc_f32(float f) {
  unsigned b = __float_as_uint(f);
  return (b & 0x80000000u) ? ~b : (b | 0x80000000u);
}
__device__ __forceinline__ float dec_f32(unsigned u) {
  return __uint_as_float((u & 0x80000000u) ? (u & 0x7FFFFFFFu) : ~u);
}

__device__ __forceinline__ float atanh_clip(float n) {
  float z = fminf(n, 1.0f - 1e-7f);
  return 0.5f * logf((1.0f + z) / (1.0f - z));
}

// ---------------- K1: logx = logmap0(x) ----------------

__global__ void __launch_bounds__(256) k_logmap0(const float* __restrict__ x,
                                                 float* __restrict__ logx, int N) {
  int node = blockIdx.x, t = threadIdx.x;
  float v = x[(size_t)node * 256 + t];
  float n = sqrtf(block_sum_256(v * v));
  n = fmaxf(n, 1e-15f);
  logx[(size_t)node * 256 + t] = (atanh_clip(n) / n) * v;
}

// ---------------- K2: WMMA GEMM  mm = logx @ W^T ----------------
// D[m,o] = sum_k A[m,k] * W[o,k].  A fragment: lane<16 row m=lane holds
// K {k0..k0+7, k0+16..k0+23}; lane>=16 same row, K offset +8 (ISA 16-bit A layout).
// B fragment mirrors it with lanes striding the 16 output columns.

#define PACK4(dst, base, f4)                                              \
  dst[(base)+0] = (_Float16)(f4).x; dst[(base)+1] = (_Float16)(f4).y;     \
  dst[(base)+2] = (_Float16)(f4).z; dst[(base)+3] = (_Float16)(f4).w;

__global__ void __launch_bounds__(128) k_gemm_wmma(const float* __restrict__ A,
                                                   const float* __restrict__ W,
                                                   float* __restrict__ Out, int N) {
  const int lane = threadIdx.x & 31;
  const int wave = threadIdx.x >> 5;
  const int rowT = blockIdx.x * 16;
  const int colT = blockIdx.y * 64 + wave * 16;
  const int half = lane >> 4;
  const int l15  = lane & 15;

  int m = rowT + l15; if (m >= N) m = N - 1;
  const int o    = colT + l15;
  const int koff = half * 8;

  const float* arow = A + (size_t)m * 256;
  const float* wrow = W + (size_t)o * 256;

  v8f acc = {};
  #pragma unroll
  for (int k0 = 0; k0 < 256; k0 += 32) {
    const float4* pa = (const float4*)(arow + k0 + koff);
    const float4* pb = (const float4*)(wrow + k0 + koff);
    float4 a0 = pa[0], a1 = pa[1], a2 = pa[4], a3 = pa[5];
    float4 b0 = pb[0], b1 = pb[1], b2 = pb[4], b3 = pb[5];
    v16h av, bv;
    PACK4(av, 0, a0)  PACK4(av, 4, a1)  PACK4(av, 8, a2)  PACK4(av, 12, a3)
    PACK4(bv, 0, b0)  PACK4(bv, 4, b1)  PACK4(bv, 8, b2)  PACK4(bv, 12, b3)
    acc = __builtin_amdgcn_wmma_f32_16x16x32_f16(
        /*neg_a=*/false, av, /*neg_b=*/false, bv,
        /*c_mod=*/(short)0, acc, /*reuse_a=*/false, /*reuse_b=*/false);
  }
  // C/D layout: VGPR r -> (M = r + half*8, N = l15)
  #pragma unroll
  for (int r = 0; r < 8; ++r) {
    int row = rowT + r + half * 8;
    if (row < N) Out[(size_t)row * 256 + (colT + l15)] = acc[r];
  }
}

// ---------------- K3: xh=proj(expmap0(mm)); xh=proj(mobius(xh,hb)); L=logmap0(xh) ----

__global__ void __launch_bounds__(256) k_hyp_bias(const float* __restrict__ mm,
                                                  const float* __restrict__ b_lin,
                                                  float* __restrict__ L, int N) {
  int node = blockIdx.x, t = threadIdx.x;
  float u  = mm[(size_t)node * 256 + t];
  float un = fmaxf(sqrtf(block_sum_256(u * u)), 1e-15f);
  float xh = (tanhf(un) / un) * u;              // expmap0
  float nh = fmaxf(tanhf(un), 1e-15f);          // |xh| (clipped)
  float ps = (nh > MAXN_PROJ) ? (MAXN_PROJ / nh) : 1.0f;
  xh *= ps;
  float x2 = nh * ps; x2 *= x2;

  float bt = b_lin[t];
  float bn = fmaxf(sqrtf(block_sum_256(bt * bt)), 1e-15f);
  float hb = (tanhf(bn) / bn) * bt;
  float nb = fmaxf(tanhf(bn), 1e-15f);
  float pb = (nb > MAXN_PROJ) ? (MAXN_PROJ / nb) : 1.0f;
  hb *= pb;
  float y2 = nb * pb; y2 *= y2;

  float xy  = block_sum_256(xh * hb);
  float num = (1.0f + 2.0f * xy + y2) * xh + (1.0f - x2) * hb;
  float den = fmaxf(1.0f + 2.0f * xy + x2 * y2, 1e-15f);
  float w   = num / den;

  float wn = fmaxf(sqrtf(block_sum_256(w * w)), 1e-15f);
  float pw = (wn > MAXN_PROJ) ? (MAXN_PROJ / wn) : 1.0f;
  w *= pw;
  float m = fmaxf(wn * pw, 1e-15f);
  L[(size_t)node * 256 + t] = (atanh_clip(m) / m) * w;  // logmap0
}

// ---------------- K4: per-(n,h) scores from the [H,N,DH] flat view of L -------

__global__ void __launch_bounds__(256) k_scores(const float* __restrict__ L,
                                                const float* __restrict__ att,
                                                float* __restrict__ s_i,
                                                float* __restrict__ s_j, int N) {
  __shared__ float ri[256], rj[256];
  int t = threadIdx.x, d = t & 63;
  long p = (long)blockIdx.x * 4 + (t >> 6);   // pair index = h*N + n
  long tot = (long)4 * N;
  float v = 0.f, ai = 0.f, aj = 0.f;
  int h = 0, n = 0;
  if (p < tot) {
    h = (int)(p / N); n = (int)(p % N);
    v  = L[p * 64 + d];
    ai = att[h * 128 + d];
    aj = att[h * 128 + 64 + d];
  }
  ri[t] = v * ai; rj[t] = v * aj; __syncthreads();
  #pragma unroll
  for (int s = 32; s > 0; s >>= 1) {
    if (d < s) { ri[t] += ri[t + s]; rj[t] += rj[t + s]; }
    __syncthreads();
  }
  if (d == 0 && p < tot) {
    s_i[(size_t)n * 4 + h] = ri[t];
    s_j[(size_t)n * 4 + h] = rj[t];
  }
}

// ---------------- K5: init accumulators ----------------

__global__ void k_init(float* __restrict__ acc, unsigned* __restrict__ segmax,
                       float* __restrict__ segsum, int N) {
  long i = (long)blockIdx.x * 256 + threadIdx.x;
  if (i < (long)N * 256) acc[i] = 0.0f;
  if (i < (long)N * 4) { segmax[i] = 0u; segsum[i] = 0.0f; }
}

// ---------------- K6: segment max over edges ----------------

__device__ __forceinline__ float lrelu(float a) { return a >= 0.f ? a : 0.2f * a; }

__global__ void k_edge_max(const int* __restrict__ ei, const float* __restrict__ si,
                           const float* __restrict__ sj, unsigned* __restrict__ segmax,
                           int E, int N) {
  long e = (long)blockIdx.x * blockDim.x + threadIdx.x;
  if (e >= (long)E + N) return;
  int s, d;
  if (e < E) { s = ei[e]; d = ei[(long)E + e]; } else { s = d = (int)(e - E); }
  float4 a = ((const float4*)si)[d];
  float4 b = ((const float4*)sj)[s];
  atomicMax(&segmax[(size_t)d * 4 + 0], enc_f32(lrelu(a.x + b.x)));
  atomicMax(&segmax[(size_t)d * 4 + 1], enc_f32(lrelu(a.y + b.y)));
  atomicMax(&segmax[(size_t)d * 4 + 2], enc_f32(lrelu(a.z + b.z)));
  atomicMax(&segmax[(size_t)d * 4 + 3], enc_f32(lrelu(a.w + b.w)));
}

// ---------------- K7: segment sum of exp(alpha - max) ----------------

__global__ void k_edge_sum(const int* __restrict__ ei, const float* __restrict__ si,
                           const float* __restrict__ sj, const unsigned* __restrict__ segmax,
                           float* __restrict__ segsum, int E, int N) {
  long e = (long)blockIdx.x * blockDim.x + threadIdx.x;
  if (e >= (long)E + N) return;
  int s, d;
  if (e < E) { s = ei[e]; d = ei[(long)E + e]; } else { s = d = (int)(e - E); }
  float4 a = ((const float4*)si)[d];
  float4 b = ((const float4*)sj)[s];
  float al[4] = { lrelu(a.x + b.x), lrelu(a.y + b.y), lrelu(a.z + b.z), lrelu(a.w + b.w) };
  #pragma unroll
  for (int h = 0; h < 4; ++h) {
    float mx = dec_f32(segmax[(size_t)d * 4 + h]);
    atomicAdd(&segsum[(size_t)d * 4 + h], expf(al[h] - mx));
  }
}

// ---------------- K8: weighted message scatter (one block per edge) ----------

__global__ void __launch_bounds__(256) k_msg(const int* __restrict__ ei,
                                             const float* __restrict__ si,
                                             const float* __restrict__ sj,
                                             const unsigned* __restrict__ segmax,
                                             const float* __restrict__ segsum,
                                             const float* __restrict__ L,
                                             float* __restrict__ acc, int E, int N) {
  long e = blockIdx.x;
  int s, d;
  if (e < E) { s = ei[e]; d = ei[(long)E + e]; } else { s = d = (int)(e - E); }
  int t = threadIdx.x, h = t >> 6, dd = t & 63;
  float al = lrelu(si[(size_t)d * 4 + h] + sj[(size_t)s * 4 + h]);
  float mx = dec_f32(segmax[(size_t)d * 4 + h]);
  float sm = fmaxf(segsum[(size_t)d * 4 + h], 1e-15f);
  float w  = expf(al - mx) / sm;
  // source feature from the [H,N,DH] flat view of L; acc is [N,H,DH]
  float v = L[((size_t)h * N + s) * 64 + dd];
  atomicAdd(&acc[(size_t)d * 256 + t], w * v);
}

// ---------------- K9: mixed-row finalize: relu -> expmap0 -> proj ------------
// mix flat f = h*N*64 + n*64 + d  maps to acc[n*256 + h*64 + d]

__global__ void __launch_bounds__(256) k_final(const float* __restrict__ acc,
                                               const float* __restrict__ b_conv,
                                               float* __restrict__ out, int N) {
  int r = blockIdx.x, c = threadIdx.x;
  size_t f  = (size_t)r * 256 + c;
  size_t nd = (size_t)N * 64;
  int h  = (int)(f / nd);
  int nn = (int)((f / 64) % N);
  int dd = (int)(f & 63);
  float v = acc[(size_t)nn * 256 + (size_t)h * 64 + dd] + b_conv[c];
  v = fmaxf(v, 0.0f);
  float vn = fmaxf(sqrtf(block_sum_256(v * v)), 1e-15f);
  float e  = (tanhf(vn) / vn) * v;
  float ne = fmaxf(tanhf(vn), 1e-15f);
  if (ne > MAXN_PROJ) e *= MAXN_PROJ / ne;
  out[f] = e;
}

// ---------------- launch ----------------

extern "C" void kernel_launch(void* const* d_in, const int* in_sizes, int n_in,
                              void* d_out, int out_size, void* d_ws, size_t ws_size,
                              hipStream_t stream) {
  const float* x      = (const float*)d_in[0];
  const int*   ei     = (const int*)d_in[1];
  const float* W      = (const float*)d_in[2];
  const float* b_lin  = (const float*)d_in[3];
  const float* att    = (const float*)d_in[4];
  const float* b_conv = (const float*)d_in[5];
  float* out = (float*)d_out;

  const int N = in_sizes[0] / 256;
  const int E = in_sizes[1] / 2;
  if (N <= 0) return;

  // workspace partition (floats)
  float* buf0 = (float*)d_ws;                    // logx, later L (logmap0(xh))
  float* buf1 = buf0 + (size_t)N * 256;          // mm, later acc
  float* s_i  = buf1 + (size_t)N * 256;          // [N,4]
  float* s_j  = s_i + (size_t)N * 4;
  unsigned* segmax = (unsigned*)(s_j + (size_t)N * 4);
  float* segsum    = (float*)(segmax + (size_t)N * 4);

  const long Etot = (long)E + N;

  k_logmap0<<<N, 256, 0, stream>>>(x, buf0, N);

  dim3 g2((N + 15) / 16, 4);
  k_gemm_wmma<<<g2, 128, 0, stream>>>(buf0, W, buf1, N);

  k_hyp_bias<<<N, 256, 0, stream>>>(buf1, b_lin, buf0, N);   // buf0 := L

  int gs = (int)(((long)4 * N + 3) / 4);
  k_scores<<<gs, 256, 0, stream>>>(buf0, att, s_i, s_j, N);

  int gi = (int)(((long)N * 256 + 255) / 256);
  k_init<<<gi, 256, 0, stream>>>(buf1, segmax, segsum, N);   // buf1 := acc

  int ge = (int)((Etot + 255) / 256);
  k_edge_max<<<ge, 256, 0, stream>>>(ei, s_i, s_j, segmax, E, N);
  k_edge_sum<<<ge, 256, 0, stream>>>(ei, s_i, s_j, segmax, segsum, E, N);

  k_msg<<<(int)Etot, 256, 0, stream>>>(ei, s_i, s_j, segmax, segsum, buf0, buf1, E, N);

  k_final<<<N, 256, 0, stream>>>(buf1, b_conv, out, N);
}
